// AddPoolingFusion_82446192214446
// MI455X (gfx1250) — matrix operand order repaired and checked by reading, not compiled, not run on hardware
//
#include <hip/hip_runtime.h>

// ---- problem constants (from reference setup_inputs) ----
#define BB 16
#define L1 2048
#define L2 2048
#define DD 768            // d1 == d2 == d3 == 768

typedef __attribute__((ext_vector_type(16))) __bf16 v16bf;
typedef __attribute__((ext_vector_type(8)))  float  v8f;
typedef __attribute__((ext_vector_type(4)))  float  v4f;

// --------------------------------------------------------------------------
// Kernel 1: convert W1 (f32, [768,768] row-major) to bf16 (RNE via cast)
// --------------------------------------------------------------------------
__global__ __launch_bounds__(256) void cvt_w1_kernel(
    const float* __restrict__ W1, unsigned short* __restrict__ Wb)
{
    int i = blockIdx.x * 256 + threadIdx.x;   // grid covers 768*768 exactly
    __bf16 h = (__bf16)W1[i];
    Wb[i] = *(const unsigned short*)&h;
}

// --------------------------------------------------------------------------
// Kernel 2: partial column sums of x2 over l2.
// grid = (3 dchunks, 16 jchunks, 16 batches), block = 256.
// part[b][jc][d] = sum over 128 rows. x2 is read exactly once -> NT loads.
// --------------------------------------------------------------------------
__global__ __launch_bounds__(256) void x2_partial_kernel(
    const float* __restrict__ x2, float* __restrict__ part)
{
    const int d  = blockIdx.x * 256 + threadIdx.x;
    const int jc = blockIdx.y;
    const int b  = blockIdx.z;
    const float* p = x2 + ((size_t)b * L2 + (size_t)jc * 128) * DD + d;
    float s0 = 0.f, s1 = 0.f, s2 = 0.f, s3 = 0.f;
    #pragma unroll 4
    for (int j = 0; j < 128; j += 4) {
        s0 += __builtin_nontemporal_load(p + (size_t)(j + 0) * DD);
        s1 += __builtin_nontemporal_load(p + (size_t)(j + 1) * DD);
        s2 += __builtin_nontemporal_load(p + (size_t)(j + 2) * DD);
        s3 += __builtin_nontemporal_load(p + (size_t)(j + 3) * DD);
    }
    part[((size_t)b * 16 + jc) * DD + d] = (s0 + s1) + (s2 + s3);
}

// --------------------------------------------------------------------------
// Kernel 3: comb[b][e] = b1[e] + b2[e] + dot(W2[e,:], mean_j x2[b,j,:])
// grid = (3 echunks, 16 batches), block = 256. xbar staged in LDS.
// --------------------------------------------------------------------------
__global__ __launch_bounds__(256) void combine_kernel(
    const float* __restrict__ part, const float* __restrict__ W2,
    const float* __restrict__ b1, const float* __restrict__ b2,
    float* __restrict__ comb)
{
    __shared__ float xbar[DD];
    const int ec = blockIdx.x;
    const int b  = blockIdx.y;

    for (int d = threadIdx.x; d < DD; d += 256) {
        float s = 0.f;
        #pragma unroll
        for (int jc = 0; jc < 16; ++jc)
            s += part[((size_t)b * 16 + jc) * DD + d];
        xbar[d] = s * (1.0f / (float)L2);
    }
    __syncthreads();

    const int e = ec * 256 + threadIdx.x;
    const float* wrow = W2 + (size_t)e * DD;
    float s = 0.f;
    for (int d = 0; d < DD; d += 4) {
        s += wrow[d + 0] * xbar[d + 0];
        s += wrow[d + 1] * xbar[d + 1];
        s += wrow[d + 2] * xbar[d + 2];
        s += wrow[d + 3] * xbar[d + 3];
    }
    comb[(size_t)b * DD + e] = s + b1[e] + b2[e];
}

// --------------------------------------------------------------------------
// Kernel 4: out[r,e] = sum_d x1[r,d]*W1[e,d] + comb[b(r), e]
// M = 32768 rows, N = 768, K = 768. bf16 WMMA 16x16x32, f32 accumulate.
// Workgroup tile 128x128, 8 waves as 4(m) x 2(n); wave tile 32x64 =
// 2x4 WMMA accumulators. grid = (6 nblk, 256 mblk), block = 256.
// --------------------------------------------------------------------------
__global__ __launch_bounds__(256) void gemm_wmma_kernel(
    const float* __restrict__ x1, const unsigned short* __restrict__ Wb,
    const float* __restrict__ comb, float* __restrict__ out)
{
    const int lane = threadIdx.x & 31;
    const int wv   = threadIdx.x >> 5;       // 0..7
    const int wm   = wv & 3;                 // 0..3
    const int wn   = wv >> 2;                // 0..1
    const int mrow0 = blockIdx.y * 128 + wm * 32;
    const int ncol0 = blockIdx.x * 128 + wn * 64;
    const int lm  = lane & 15;               // 0..15
    const int grp = lane >> 4;               // 0..1

    const v8f vzero = {0.f, 0.f, 0.f, 0.f, 0.f, 0.f, 0.f, 0.f};
    v8f acc[2][4];
    #pragma unroll
    for (int mt = 0; mt < 2; ++mt) {
        #pragma unroll
        for (int nt = 0; nt < 4; ++nt) acc[mt][nt] = vzero;
    }

    for (int k0 = 0; k0 < DD; k0 += 32) {
        // ---- A fragments: 16x32 bf16 per m-tile, converted from f32 ----
        // contiguous k order (same as B): lane holds row m = lane%16; K = grp*16 + 0..15
        v16bf afrag[2];
        #pragma unroll
        for (int mt = 0; mt < 2; ++mt) {
            const float* ap =
                x1 + (size_t)(mrow0 + mt * 16 + lm) * DD + k0 + grp * 16;
            v4f f0 = *(const v4f*)(ap + 0);
            v4f f1 = *(const v4f*)(ap + 4);
            v4f f2 = *(const v4f*)(ap + 8);
            v4f f3 = *(const v4f*)(ap + 12);
            v16bf a;
            #pragma unroll
            for (int t = 0; t < 4; ++t) a[t]      = (__bf16)f0[t];
            #pragma unroll
            for (int t = 0; t < 4; ++t) a[4 + t]  = (__bf16)f1[t];
            #pragma unroll
            for (int t = 0; t < 4; ++t) a[8 + t]  = (__bf16)f2[t];
            #pragma unroll
            for (int t = 0; t < 4; ++t) a[12 + t] = (__bf16)f3[t];
            afrag[mt] = a;
        }
        // ---- B fragments: 32x16 bf16 per n-tile, straight from bf16 W1 ----
        // lane holds col n = lane%16; K = grp*16 + 0..15 -> 32B contiguous
        v16bf bfrag[4];
        #pragma unroll
        for (int nt = 0; nt < 4; ++nt) {
            const unsigned short* bp =
                Wb + (size_t)(ncol0 + nt * 16 + lm) * DD + k0 + grp * 16;
            bfrag[nt] = *(const v16bf*)bp;
        }
        // ---- 8 WMMAs ----
        #pragma unroll
        for (int mt = 0; mt < 2; ++mt) {
            #pragma unroll
            for (int nt = 0; nt < 4; ++nt) {
                acc[mt][nt] = __builtin_amdgcn_wmma_f32_16x16x32_bf16(
                    false, afrag[mt], false, bfrag[nt],
                    (short)0, acc[mt][nt], false, false);
            }
        }
    }

    // ---- Epilogue: add fused bias, NT store (out is write-once) ----
    // C/D layout: lane holds col n = lane%16; VGPR v holds row grp*8 + v.
    const int bidx = (int)blockIdx.y / (L1 / 128);  // batch of this m-block
    #pragma unroll
    for (int nt = 0; nt < 4; ++nt) {
        const int col = ncol0 + nt * 16 + lm;
        const float cb = comb[(size_t)bidx * DD + col];
        #pragma unroll
        for (int mt = 0; mt < 2; ++mt) {
            #pragma unroll
            for (int v = 0; v < 8; ++v) {
                const int row = mrow0 + mt * 16 + grp * 8 + v;
                __builtin_nontemporal_store(acc[mt][nt][v] + cb,
                                            out + (size_t)row * DD + col);
            }
        }
    }
}

// --------------------------------------------------------------------------
extern "C" void kernel_launch(void* const* d_in, const int* in_sizes, int n_in,
                              void* d_out, int out_size, void* d_ws, size_t ws_size,
                              hipStream_t stream) {
    (void)in_sizes; (void)n_in; (void)out_size; (void)ws_size;
    const float* x1 = (const float*)d_in[0];   // [16, 2048, 768]
    const float* x2 = (const float*)d_in[1];   // [16, 2048, 768]
    const float* W1 = (const float*)d_in[2];   // [768, 768]
    const float* b1 = (const float*)d_in[3];   // [768]
    const float* W2 = (const float*)d_in[4];   // [768, 768]
    const float* b2 = (const float*)d_in[5];   // [768]
    float* out = (float*)d_out;                // [16, 2048, 768]

    char* ws = (char*)d_ws;
    unsigned short* Wb = (unsigned short*)ws;            // 768*768*2 = 1179648 B
    float* part = (float*)(ws + 1179648);                // 16*16*768*4 = 786432 B
    float* comb = (float*)(ws + 1179648 + 786432);       // 16*768*4   =  49152 B

    cvt_w1_kernel<<<dim3((DD * DD) / 256), 256, 0, stream>>>(W1, Wb);
    x2_partial_kernel<<<dim3(3, 16, 16), 256, 0, stream>>>(x2, part);
    combine_kernel<<<dim3(3, 16), 256, 0, stream>>>(part, W2, b1, b2, comb);
    gemm_wmma_kernel<<<dim3(6, 256), 256, 0, stream>>>(x1, Wb, comb, out);
}
